// YoshidaIntegrator_23210003268311
// MI455X (gfx1250) — matrix operand order repaired and checked by reading, not compiled, not run on hardware
//
#include <hip/hip_runtime.h>

// ---------------------------------------------------------------------------
// Yoshida 4th-order integrator with low-rank Christoffel term, fully fused.
// B=4096 rows, D=1024, R=256.  All 6 GEMMs done with v_wmma_f32_16x16x32_f16.
// One workgroup (256 threads = 8 wave32) owns BM=16 batch rows for the whole
// integration; v lives in LDS (f32 master + f16 WMMA copy), x-increment
// accumulates in C-fragment registers.  U/W converted once to f16 in d_ws.
// ---------------------------------------------------------------------------

typedef _Float16 h8   __attribute__((ext_vector_type(8)));   // 16B chunk
typedef _Float16 v16h __attribute__((ext_vector_type(16)));  // WMMA A/B frag
typedef float    v8f  __attribute__((ext_vector_type(8)));   // WMMA C/D frag

#define Bdim 4096
#define Ddim 1024
#define Rdim 256
#define BM   16      // batch rows per workgroup

// --- f32 -> f16 conversion of U [R,D] and W [D,R] into workspace ------------
__global__ void yoshida_cvt_kernel(const float* __restrict__ U,
                                   const float* __restrict__ W,
                                   _Float16* __restrict__ Uh,
                                   _Float16* __restrict__ Wh, int n)
{
    int i = blockIdx.x * blockDim.x + threadIdx.x;
    if (i < n) {
        Uh[i] = (_Float16)U[i];
        Wh[i] = (_Float16)W[i];
    }
}

__global__ __launch_bounds__(256, 1)
void yoshida_main_kernel(const float* __restrict__ xg,
                         const float* __restrict__ vg,
                         const float* __restrict__ fg,
                         const _Float16* __restrict__ Uh,   // [R][D] row-major f16
                         const _Float16* __restrict__ Wh,   // [D][R] row-major f16
                         float* __restrict__ out_x,
                         float* __restrict__ out_v)
{
    constexpr int DP = Ddim + 8;   // padded f16 row (bank-conflict-free A-frag)
    constexpr int RP = Rdim + 8;

    __shared__ float    Vf[BM][Ddim];   // master velocity (f32)      64.0 KB
    __shared__ _Float16 Vh[BM][DP];     // WMMA copy of velocity      32.25 KB
    __shared__ _Float16 Sh[BM][RP];     // s = (U v)^2, f16            8.25 KB

    const int tid  = threadIdx.x;
    const int wave = tid >> 5;          // 0..7
    const int lane = tid & 31;
    const int hi   = lane >> 4;         // half-wave select
    const int l16  = lane & 15;
    const long b0  = (long)blockIdx.x * BM;

    // Yoshida coefficients (compile-time)
    constexpr double CBRT2 = 1.2599210498948731647672106072782;
    constexpr double W1d   =  1.0 / (2.0 - CBRT2);
    constexpr double W0d   = -CBRT2 / (2.0 - CBRT2);
    constexpr float  dt    = 0.01f;     // DT * DT_SCALE
    const float Dstep[3] = { (float)W1d, (float)W0d, (float)W1d };
    const float Cacc[4]  = { (float)(W1d * 0.5), (float)((W0d + W1d) * 0.5),
                             (float)((W0d + W1d) * 0.5), (float)(W1d * 0.5) };

    // ---- stage v tile into LDS (f32 + f16) --------------------------------
    for (int i = tid; i < BM * Ddim; i += 256) {
        const int m = i >> 10;              // Ddim == 1024
        const int d = i & (Ddim - 1);
        const float vv = vg[(b0 + m) * Ddim + d];
        Vf[m][d] = vv;
        Vh[m][d] = (_Float16)vv;
    }
    __syncthreads();

    // ---- x-increment fragments: xacc = C1*v0 (+C2*v1 +C3*v2 +C4*v3 later) --
    // Each wave owns 8 contiguous 16-wide D-tiles (C-fragment layout).
    v8f xacc[8];
#pragma unroll
    for (int i = 0; i < 8; ++i) {
        const int d0 = (wave * 8 + i) * 16;
#pragma unroll
        for (int j = 0; j < 8; ++j)
            xacc[i][j] = Cacc[0] * Vf[j + 8 * hi][d0 + l16];
    }

#pragma unroll
    for (int s = 0; s < 3; ++s) {
        // ======== GEMM1:  H = V · Uᵀ  ;  Sh = (H∘H) in f16 ==================
        {
            const int r0a = (wave * 2 + 0) * 16;
            const int r0b = (wave * 2 + 1) * 16;
            v8f acc0 = {}; v8f acc1 = {};
            const _Float16* va = &Vh[l16][0];                          // LDS
            const _Float16* ua = Uh + (long)(r0a + l16) * Ddim + 8 * hi;
            const _Float16* ub = Uh + (long)(r0b + l16) * Ddim + 8 * hi;
#pragma unroll 4
            for (int kk = 0; kk < Ddim; kk += 32) {
                v16h A;
                ((h8*)&A)[0] = *(const h8*)(va + kk + 8 * hi);
                ((h8*)&A)[1] = *(const h8*)(va + kk + 16 + 8 * hi);
                v16h B0;
                ((h8*)&B0)[0] = *(const h8*)(ua + kk);
                ((h8*)&B0)[1] = *(const h8*)(ua + kk + 16);
                v16h B1;
                ((h8*)&B1)[0] = *(const h8*)(ub + kk);
                ((h8*)&B1)[1] = *(const h8*)(ub + kk + 16);
                acc0 = __builtin_amdgcn_wmma_f32_16x16x32_f16(
                           false, A, false, B0, (short)0, acc0, false, false);
                acc1 = __builtin_amdgcn_wmma_f32_16x16x32_f16(
                           false, A, false, B1, (short)0, acc1, false, false);
            }
#pragma unroll
            for (int j = 0; j < 8; ++j) {
                Sh[j + 8 * hi][r0a + l16] = (_Float16)(acc0[j] * acc0[j]);
                Sh[j + 8 * hi][r0b + l16] = (_Float16)(acc1[j] * acc1[j]);
            }
        }
        __syncthreads();   // Sh ready; all Vh reads of this stage done

        // ======== GEMM2:  Γ = S · Wᵀ  ;  v += D_s·dt·(force − Γ) ===========
#pragma unroll 1
        for (int i = 0; i < 8; ++i) {
            const int d0 = (wave * 8 + i) * 16;
            v8f acc = {};
            const _Float16* sa = &Sh[l16][0];                          // LDS
            const _Float16* wb = Wh + (long)(d0 + l16) * Rdim + 8 * hi;
#pragma unroll
            for (int rr = 0; rr < Rdim; rr += 32) {
                v16h A;
                ((h8*)&A)[0] = *(const h8*)(sa + rr + 8 * hi);
                ((h8*)&A)[1] = *(const h8*)(sa + rr + 16 + 8 * hi);
                v16h B;
                ((h8*)&B)[0] = *(const h8*)(wb + rr);
                ((h8*)&B)[1] = *(const h8*)(wb + rr + 16);
                acc = __builtin_amdgcn_wmma_f32_16x16x32_f16(
                          false, A, false, B, (short)0, acc, false, false);
            }
#pragma unroll
            for (int j = 0; j < 8; ++j) {
                const int m = j + 8 * hi;
                const int d = d0 + l16;
                const float fo = fg[(b0 + m) * Ddim + d];
                const float vn = Vf[m][d] + dt * Dstep[s] * (fo - acc[j]);
                Vf[m][d] = vn;
                Vh[m][d] = (_Float16)vn;
                xacc[i][j] += Cacc[s + 1] * vn;   // C2*v1 / C3*v2 / C4*v3
            }
        }
        __syncthreads();   // Vh/Vf updated for next stage's GEMM1
    }

    // ---- finalize: x_final = x + dt*xacc ; v_out = v3 ----------------------
#pragma unroll 1
    for (int i = 0; i < 8; ++i) {
        const int d0 = (wave * 8 + i) * 16;
#pragma unroll
        for (int j = 0; j < 8; ++j) {
            const int m = j + 8 * hi;
            const int d = d0 + l16;
            const long g = (b0 + m) * Ddim + d;
            out_x[g] = xg[g] + dt * xacc[i][j];
            out_v[g] = Vf[m][d];
        }
    }
}

extern "C" void kernel_launch(void* const* d_in, const int* in_sizes, int n_in,
                              void* d_out, int out_size, void* d_ws, size_t ws_size,
                              hipStream_t stream)
{
    const float* x = (const float*)d_in[0];     // [B,D]
    const float* v = (const float*)d_in[1];     // [B,D]
    const float* f = (const float*)d_in[2];     // [B,D]
    const float* U = (const float*)d_in[3];     // [R,D]
    const float* W = (const float*)d_in[4];     // [D,R]

    _Float16* Uh = (_Float16*)d_ws;                       // 512 KB
    _Float16* Wh = Uh + (size_t)Rdim * Ddim;              // 512 KB
    float* out_x = (float*)d_out;
    float* out_v = out_x + (size_t)Bdim * Ddim;

    const int n = Rdim * Ddim;                            // == Ddim*Rdim
    hipLaunchKernelGGL(yoshida_cvt_kernel, dim3((n + 255) / 256), dim3(256), 0,
                       stream, U, W, Uh, Wh, n);
    hipLaunchKernelGGL(yoshida_main_kernel, dim3(Bdim / BM), dim3(256), 0,
                       stream, x, v, f, Uh, Wh, out_x, out_v);
}
